// MultiHeadAttention_23227183137498
// MI455X (gfx1250) — matrix operand order repaired and checked by reading, not compiled, not run on hardware
//
#include <hip/hip_runtime.h>
#include <hip/hip_bf16.h>

// ---------------------------------------------------------------------------
// MI455X (gfx1250) pair-biased MHA, wave32. All matmuls on the bf16 WMMA
// pipe (v_wmma_f32_16x16x32_bf16, fp32 accumulate). All WMMA operands are
// pre-staged in bf16 in fragment-native layouts so every fragment load is a
// contiguous global_load_b128 (no strided scalar loads, no serializing cvt
// chains). Embedding tables staged to LDS via global_load_async_to_lds_b128
// (ASYNCcnt path). Scores/softmax operate directly on the attn region of
// d_out to avoid an extra 67MB round trip.
// ---------------------------------------------------------------------------

typedef __attribute__((ext_vector_type(16))) __bf16 v16bf;
typedef __attribute__((ext_vector_type(8)))  __bf16 v8bf;
typedef __attribute__((ext_vector_type(4)))  __bf16 v4bf;
typedef __attribute__((ext_vector_type(8)))  float  v8f;

#define WMMA_BF16(a, b, c) \
  __builtin_amdgcn_wmma_f32_16x16x32_bf16(false, (a), false, (b), (short)0, (c), false, false)

static constexpr int kL = 1024;   // sequence length
static constexpr int kD = 512;    // d_model = H * d_k

// ---- fragment loaders (layouts per CDNA5 ISA 7.12.2, wave32) --------------
// 16-bit A 16x32 (MxK), bf16 row-major storage: lane l holds row l&15;
// half = l>>4 selects K base half*8 (v0..3) and 16+half*8 (v4..7); each
// group of 8 K-values is contiguous -> two b128 loads.
static __device__ inline v16bf load_a_frag_bf(const __bf16* __restrict__ A, int lda) {
  const int lane = threadIdx.x & 31;
  const int r = lane & 15, half = lane >> 4;
  const __bf16* row = A + r * lda + (half << 3);
  const v8bf lo = *(const v8bf*)(row);        // K = half*8 + 0..7
  const v8bf hi = *(const v8bf*)(row + 16);   // K = 16 + half*8 + 0..7
  return __builtin_shufflevector(lo, hi, 0, 1, 2, 3, 4, 5, 6, 7,
                                 8, 9, 10, 11, 12, 13, 14, 15);
}

// Same A fragment but from f32 row-major (attn matrix): contiguous float4
// loads + packed cvt.
static __device__ inline v16bf load_a_frag_f32(const float* __restrict__ A, int lda) {
  const int lane = threadIdx.x & 31;
  const int r = lane & 15, half = lane >> 4;
  const float* row = A + r * lda + (half << 3);
  const float4 p0 = *(const float4*)(row);
  const float4 p1 = *(const float4*)(row + 4);
  const float4 p2 = *(const float4*)(row + 16);
  const float4 p3 = *(const float4*)(row + 20);
  v16bf a;
  a[0]  = (__bf16)p0.x; a[1]  = (__bf16)p0.y; a[2]  = (__bf16)p0.z; a[3]  = (__bf16)p0.w;
  a[4]  = (__bf16)p1.x; a[5]  = (__bf16)p1.y; a[6]  = (__bf16)p1.z; a[7]  = (__bf16)p1.w;
  a[8]  = (__bf16)p2.x; a[9]  = (__bf16)p2.y; a[10] = (__bf16)p2.z; a[11] = (__bf16)p2.w;
  a[12] = (__bf16)p3.x; a[13] = (__bf16)p3.y; a[14] = (__bf16)p3.z; a[15] = (__bf16)p3.w;
  return a;
}

// 16-bit B 32x16 (KxN) from transposed bf16 storage Bt[c][k] (ld = ldbt):
// lane l: col c = l&15; half = l>>4 -> K = half*8 + 0..7 (v0..3) and
// 16 + half*8 + 0..7 (v4..7), matching the A fragment -> two b128 loads.
static __device__ inline v16bf load_bT_frag_bf(const __bf16* __restrict__ Bt, int ldbt) {
  const int lane = threadIdx.x & 31;
  const int c = lane & 15, half = lane >> 4;
  const __bf16* row = Bt + c * ldbt + (half << 3);
  const v8bf lo = *(const v8bf*)(row);
  const v8bf hi = *(const v8bf*)(row + 16);
  return __builtin_shufflevector(lo, hi, 0, 1, 2, 3, 4, 5, 6, 7,
                                 8, 9, 10, 11, 12, 13, 14, 15);
}

// ---- elementwise f32 -> bf16 ----------------------------------------------
__global__ __launch_bounds__(256)
void cvt_bf16_kernel(const float* __restrict__ src, __bf16* __restrict__ dst) {
  const int i = (blockIdx.x * 256 + threadIdx.x) << 2;
  const float4 p = *(const float4*)(src + i);
  v4bf o;
  o[0] = (__bf16)p.x; o[1] = (__bf16)p.y; o[2] = (__bf16)p.z; o[3] = (__bf16)p.w;
  *(v4bf*)(dst + i) = o;
}

// ---- 512x512 transpose + convert (W[k][n] -> Wt[n][k] bf16), LDS tiled ----
__global__ __launch_bounds__(256)
void transpose_cvt_kernel(const float* __restrict__ w, __bf16* __restrict__ wT) {
  __shared__ float tile[16][17];
  const int tx = threadIdx.x & 15, ty = threadIdx.x >> 4;
  const int n = (blockIdx.x << 4) + tx;          // col (coalesced read)
  const int k = (blockIdx.y << 4) + ty;          // row
  tile[ty][tx] = w[k * kD + n];
  __syncthreads();
  // write wT[n][k], coalesced in k
  wT[((blockIdx.x << 4) + ty) * kD + (blockIdx.y << 4) + tx] = (__bf16)tile[tx][ty];
}

// ---- GEMM: 16x64 tile of A[MxK](bf16) * Bt[NxK](bf16) + bias --------------
// mode 0: Cf[m*N+n] = acc + bias + residual   (f32 out)
// mode 1: Cb[m*N+n] = bf16(acc + bias)        (row-major bf16 out)
// mode 2: Cb[((b*8+h)*64+d)*1024 + j]         (vh transposed bf16 out)
__global__ __launch_bounds__(32)
void gemm16x64(const __bf16* __restrict__ A, const __bf16* __restrict__ Bt,
               const float* __restrict__ bias, const float* __restrict__ residual,
               float* __restrict__ Cf, __bf16* __restrict__ Cb,
               int mode, int N, int K) {
  const int tm = blockIdx.x << 4;
  const int tn = blockIdx.y << 6;
  v8f acc[4] = {};
  for (int kk = 0; kk < K; kk += 32) {
    const v16bf a = load_a_frag_bf(A + tm * K + kk, K);
#pragma unroll
    for (int t = 0; t < 4; ++t) {
      const v16bf b = load_bT_frag_bf(Bt + (tn + (t << 4)) * K + kk, K);
      acc[t] = WMMA_BF16(a, b, acc[t]);
    }
  }
  const int lane = threadIdx.x & 31;
  const int n0 = lane & 15, half = lane >> 4;
#pragma unroll
  for (int t = 0; t < 4; ++t) {
    const int n = tn + (t << 4) + n0;
    const float bv = bias ? bias[n] : 0.0f;
#pragma unroll
    for (int v = 0; v < 8; ++v) {
      const int m = tm + v + (half << 3);
      float val = acc[t][v] + bv;
      if (mode == 0) {
        if (residual) val += residual[m * N + n];
        Cf[m * N + n] = val;
      } else if (mode == 1) {
        Cb[m * N + n] = (__bf16)val;
      } else {                                      // vh -> [b][h][d][j] bf16
        const int b = m >> 10, j = m & 1023, hh = n >> 6, d = n & 63;
        Cb[(((b << 3) + hh) * 64 + d) * kL + j] = (__bf16)val;
      }
    }
  }
}

// ---- scores: QK^T (WMMA) + 5-feature pair bias + mask -> attn (pre-softmax)
__global__ __launch_bounds__(32)
void scores_bias_kernel(const __bf16* __restrict__ qh, const __bf16* __restrict__ kh,
                        const int* __restrict__ tf1, const int* __restrict__ tf2,
                        const int* __restrict__ tf3, const int* __restrict__ tf4,
                        const int* __restrict__ tf5,
                        const float* __restrict__ e1, const float* __restrict__ e2,
                        const float* __restrict__ e3, const float* __restrict__ e4,
                        const float* __restrict__ e5,
                        const unsigned char* __restrict__ mask,
                        float* __restrict__ attn) {
  __shared__ float semb[5 * 512];        // 5 tables of [64 vocab x 8 heads]
  {
    // Async copy-engine staging: global -> LDS without passing through VGPRs.
    // Generic-pointer low 32 bits == LDS byte offset (flat->LDS uses addr[31:0]).
    const float* srcs[5] = {e1, e2, e3, e4, e5};
    const unsigned ldsbase = (unsigned)(uintptr_t)(&semb[0]);
#pragma unroll
    for (int f = 0; f < 5; ++f) {
      const float* s = srcs[f] + (threadIdx.x << 2);          // 4 floats / lane
      const unsigned l = ldsbase + f * 2048 + (threadIdx.x << 4);
#pragma unroll
      for (int it = 0; it < 4; ++it) {                        // 4 x 32 lanes x 16B = 2KB
        asm volatile("global_load_async_to_lds_b128 %0, %1, off"
                     :: "v"(l + it * 512), "v"(s + it * 128) : "memory");
      }
    }
    asm volatile("s_wait_asynccnt 0" ::: "memory");
  }
  __syncthreads();

  const int z = blockIdx.z;              // z = h*2 + b  (== attn slab h*B+b)
  const int h = z >> 1, b = z & 1;
  const int ti = blockIdx.x << 4;
  const int tj = blockIdx.y << 6;
  const int bLL = b * kL * kL;

  // speculative prefetch of the pair-feature rows this tile will gather
  __builtin_prefetch(tf1 + bLL + (ti + (threadIdx.x & 15)) * kL + tj, 0, 1);

  const __bf16* Aq = qh + (b * kL + ti) * kD + (h << 6);
  v8f acc[4] = {};
#pragma unroll
  for (int kk = 0; kk < 64; kk += 32) {
    const v16bf a = load_a_frag_bf(Aq + kk, kD);
#pragma unroll
    for (int t = 0; t < 4; ++t) {
      // B(dk, j) = kh[b, j, h*64+dk]; contiguous in dk -> bT layout, ld=512
      const v16bf bb = load_bT_frag_bf(kh + (b * kL + tj + (t << 4)) * kD + (h << 6) + kk, kD);
      acc[t] = WMMA_BF16(a, bb, acc[t]);
    }
  }

  const int lane = threadIdx.x & 31;
  const int n0 = lane & 15, half = lane >> 4;
#pragma unroll
  for (int t = 0; t < 4; ++t) {
    const int j = tj + (t << 4) + n0;
#pragma unroll
    for (int v = 0; v < 8; ++v) {
      const int i = ti + v + (half << 3);
      const int p = bLL + i * kL + j;
      float s = acc[t][v];
      const int t1 = tf1[p];
      if (t1 >= 0) {                     // tf_mask from tf_f1 sign gates all 5
        s += semb[(t1 << 3) + h]
           + semb[512  + (tf2[p] << 3) + h]
           + semb[1024 + (tf3[p] << 3) + h]
           + semb[1536 + (tf4[p] << 3) + h]
           + semb[2048 + (tf5[p] << 3) + h];
      }
      s *= 0.125f;                       // 1 / sqrt(64)
      if (mask[p]) s = -__builtin_inff();
      attn[(z * kL + i) * kL + j] = s;
    }
  }
}

// ---- in-place row softmax over attn[16384][1024] --------------------------
__global__ __launch_bounds__(256)
void softmax_row_kernel(float* __restrict__ attn) {
  float* p = attn + (size_t)blockIdx.x * kL;
  const int t = threadIdx.x;
  float x0 = p[t], x1 = p[t + 256], x2 = p[t + 512], x3 = p[t + 768];
  float m = fmaxf(fmaxf(x0, x1), fmaxf(x2, x3));
#pragma unroll
  for (int off = 16; off; off >>= 1) m = fmaxf(m, __shfl_xor(m, off, 32));
  __shared__ float rmax[8], rsum[8];
  if ((t & 31) == 0) rmax[t >> 5] = m;
  __syncthreads();
  float gm = rmax[0];
#pragma unroll
  for (int i = 1; i < 8; ++i) gm = fmaxf(gm, rmax[i]);
  x0 = __expf(x0 - gm); x1 = __expf(x1 - gm);
  x2 = __expf(x2 - gm); x3 = __expf(x3 - gm);
  float s = x0 + x1 + x2 + x3;
#pragma unroll
  for (int off = 16; off; off >>= 1) s += __shfl_xor(s, off, 32);
  if ((t & 31) == 0) rsum[t >> 5] = s;
  __syncthreads();
  float gs = 0.0f;
#pragma unroll
  for (int i = 0; i < 8; ++i) gs += rsum[i];
  const float inv = 1.0f / gs;
  p[t] = x0 * inv; p[t + 256] = x1 * inv; p[t + 512] = x2 * inv; p[t + 768] = x3 * inv;
}

// ---- ctx = attn * V (per b,h): A = attn f32 (on-the-fly cvt), B = vhT bf16 -
__global__ __launch_bounds__(32)
void av_kernel(const float* __restrict__ attn, const __bf16* __restrict__ vhT,
               __bf16* __restrict__ ctx) {
  const int z = blockIdx.y;              // z = h*2 + b
  const int h = z >> 1, b = z & 1;
  const int ti = blockIdx.x << 4;
  const float* A = attn + (size_t)(z * kL + ti) * kL;
  const __bf16* Vt = vhT + (((b << 3) + h) << 6) * kL;   // [64][1024], d-major
  v8f acc[4] = {};
  for (int kk = 0; kk < kL; kk += 32) {
    __builtin_prefetch(A + (threadIdx.x & 15) * kL + kk + 256, 0, 1);
    const v16bf a = load_a_frag_f32(A + kk, kL);
#pragma unroll
    for (int t = 0; t < 4; ++t) {
      // B(j, d) with d-major storage: element (k=j, c=d) = Vt[(t*16+c)*1024 + j]
      const v16bf bb = load_bT_frag_bf(Vt + (t << 4) * kL + kk, kL);
      acc[t] = WMMA_BF16(a, bb, acc[t]);
    }
  }
  const int lane = threadIdx.x & 31;
  const int n0 = lane & 15, half = lane >> 4;
#pragma unroll
  for (int t = 0; t < 4; ++t) {
    const int d = (t << 4) + n0;
#pragma unroll
    for (int v = 0; v < 8; ++v) {
      const int i = ti + v + (half << 3);
      ctx[(b * kL + i) * kD + (h << 6) + d] = (__bf16)acc[t][v];
    }
  }
}

// ---- LayerNorm over rows of 512 -------------------------------------------
__global__ __launch_bounds__(256)
void layernorm_kernel(const float* __restrict__ x, const float* __restrict__ g,
                      const float* __restrict__ beta, float* __restrict__ out) {
  const int row = blockIdx.x;
  const float* p = x + row * kD;
  const int t = threadIdx.x;
  const float a = p[t], c = p[t + 256];
  float s = a + c;
#pragma unroll
  for (int off = 16; off; off >>= 1) s += __shfl_xor(s, off, 32);
  __shared__ float r1[8], r2[8];
  if ((t & 31) == 0) r1[t >> 5] = s;
  __syncthreads();
  float tot = 0.0f;
#pragma unroll
  for (int i = 0; i < 8; ++i) tot += r1[i];
  const float mu = tot * (1.0f / 512.0f);
  const float d0 = a - mu, d1 = c - mu;
  float vs = d0 * d0 + d1 * d1;
#pragma unroll
  for (int off = 16; off; off >>= 1) vs += __shfl_xor(vs, off, 32);
  if ((t & 31) == 0) r2[t >> 5] = vs;
  __syncthreads();
  float vtot = 0.0f;
#pragma unroll
  for (int i = 0; i < 8; ++i) vtot += r2[i];
  const float inv = rsqrtf(vtot * (1.0f / 512.0f) + 1e-5f);
  out[row * kD + t]       = d0 * inv * g[t] + beta[t];
  out[row * kD + t + 256] = d1 * inv * g[t + 256] + beta[t + 256];
}

// ---------------------------------------------------------------------------
extern "C" void kernel_launch(void* const* d_in, const int* in_sizes, int n_in,
                              void* d_out, int out_size, void* d_ws, size_t ws_size,
                              hipStream_t stream) {
  const float* q   = (const float*)d_in[0];
  const float* k   = (const float*)d_in[1];
  const float* v   = (const float*)d_in[2];
  const unsigned char* mask = (const unsigned char*)d_in[3];   // jnp bool = 1 byte
  const int* tf1 = (const int*)d_in[4];
  const int* tf2 = (const int*)d_in[5];
  const int* tf3 = (const int*)d_in[6];
  const int* tf4 = (const int*)d_in[7];
  const int* tf5 = (const int*)d_in[8];
  const float* w_q = (const float*)d_in[9];
  const float* b_q = (const float*)d_in[10];
  const float* w_k = (const float*)d_in[11];
  const float* b_k = (const float*)d_in[12];
  const float* w_v = (const float*)d_in[13];
  const float* b_v = (const float*)d_in[14];
  const float* e1  = (const float*)d_in[15];
  const float* e2  = (const float*)d_in[16];
  const float* e3  = (const float*)d_in[17];
  const float* e4  = (const float*)d_in[18];
  const float* e5  = (const float*)d_in[19];
  const float* fc_w = (const float*)d_in[20];
  const float* fc_b = (const float*)d_in[21];
  const float* ln_g = (const float*)d_in[22];
  const float* ln_b = (const float*)d_in[23];

  float* out  = (float*)d_out;
  float* attn = out + (size_t)2 * kL * kD;        // 1,048,576 floats of `out` first

  // workspace layout (bytes, all 256B aligned); total = 20 MB
  char* ws = (char*)d_ws;
  __bf16* qbf   = (__bf16*)(ws);                  // 3 x 2MB input bf16
  __bf16* kbf   = (__bf16*)(ws + (2u << 20));
  __bf16* vbf   = (__bf16*)(ws + (4u << 20));
  __bf16* wqT   = (__bf16*)(ws + (6u << 20));     // 4 x 512KB transposed weights
  __bf16* wkT   = (__bf16*)(ws + (6u << 20) + (512u << 10));
  __bf16* wvT   = (__bf16*)(ws + (7u << 20));
  __bf16* fcwT  = (__bf16*)(ws + (7u << 20) + (512u << 10));
  __bf16* qh    = (__bf16*)(ws + (8u << 20));     // 2MB  [B,L,512] bf16
  __bf16* kh    = (__bf16*)(ws + (10u << 20));    // 2MB
  __bf16* vhT   = (__bf16*)(ws + (12u << 20));    // 2MB  [b][h][64][1024] bf16
  __bf16* ctxb  = (__bf16*)(ws + (14u << 20));    // 2MB  [B,L,512] bf16
  float*  fco   = (float*)(ws + (16u << 20));     // 4MB  f32 pre-LN

  // stage operands as bf16 in fragment-native layouts
  cvt_bf16_kernel<<<1024, 256, 0, stream>>>(q, qbf);
  cvt_bf16_kernel<<<1024, 256, 0, stream>>>(k, kbf);
  cvt_bf16_kernel<<<1024, 256, 0, stream>>>(v, vbf);
  const dim3 gtr(32, 32);
  transpose_cvt_kernel<<<gtr, 256, 0, stream>>>(w_q, wqT);
  transpose_cvt_kernel<<<gtr, 256, 0, stream>>>(w_k, wkT);
  transpose_cvt_kernel<<<gtr, 256, 0, stream>>>(w_v, wvT);
  transpose_cvt_kernel<<<gtr, 256, 0, stream>>>(fc_w, fcwT);

  // projections (WMMA): qh/kh row-major bf16, vh transposed bf16
  const dim3 gproj(128, 8);                       // M=2048/16, N=512/64
  gemm16x64<<<gproj, 32, 0, stream>>>(qbf, wqT, b_q, nullptr, nullptr, qh, 1, kD, kD);
  gemm16x64<<<gproj, 32, 0, stream>>>(kbf, wkT, b_k, nullptr, nullptr, kh, 1, kD, kD);
  gemm16x64<<<gproj, 32, 0, stream>>>(vbf, wvT, b_v, nullptr, nullptr, vhT, 2, kD, kD);

  // scores + pair bias + mask -> attn region of d_out (pre-softmax)
  const dim3 gsc(64, 16, 16);                     // i-tiles, j-quads, (h,b)
  scores_bias_kernel<<<gsc, 32, 0, stream>>>(qh, kh, tf1, tf2, tf3, tf4, tf5,
                                             e1, e2, e3, e4, e5, mask, attn);

  softmax_row_kernel<<<16 * kL, 256, 0, stream>>>(attn);

  const dim3 gav(64, 16);
  av_kernel<<<gav, 32, 0, stream>>>(attn, vhT, ctxb);

  // FC + bias + residual (f32 out), then LayerNorm -> d_out
  gemm16x64<<<gproj, 32, 0, stream>>>(ctxb, fcwT, fc_b, q, fco, nullptr, 0, kD, kD);
  layernorm_kernel<<<2048, 256, 0, stream>>>(fco, ln_g, ln_b, out);
}
